// ArcFaceHead_22926535426393
// MI455X (gfx1250) — matrix operand order repaired and checked by reading, not compiled, not run on hardware
//
#include <hip/hip_runtime.h>
#include <math.h>

typedef __attribute__((ext_vector_type(16))) __bf16 v16bf_t;
typedef __attribute__((ext_vector_type(8)))  float  v8f_t;
typedef __attribute__((ext_vector_type(4)))  float  f32x4_t;   // native vector for NT builtins

#define ARC_S   30.0f
#define ARC_M   0.5f
#define ARC_EPS 1e-7f

#define TN   128      // columns (classes) per block tile
#define LDB  520      // padded LDS row stride in bf16 elements (bank-conflict free)

// ---------------- kernel 1: weight row inverse norms ----------------
__global__ void arc_wnorm_kernel(const float* __restrict__ W,
                                 float* __restrict__ winv,
                                 int C, int D) {
  const int w    = threadIdx.x >> 5;
  const int lane = threadIdx.x & 31;
  const int row  = blockIdx.x * 8 + w;
  if (row >= C) return;
  const float* p = W + (size_t)row * D;
  float ss = 0.f;
  for (int k = lane; k < D; k += 32) { float v = p[k]; ss += v * v; }
  ss += __shfl_xor(ss, 16);
  ss += __shfl_xor(ss, 8);
  ss += __shfl_xor(ss, 4);
  ss += __shfl_xor(ss, 2);
  ss += __shfl_xor(ss, 1);
  if (lane == 0) winv[row] = 1.0f / fmaxf(sqrtf(ss), 1e-12f);
}

// ---------------- kernel 2: normalize embeddings -> bf16 ----------------
__global__ void arc_enorm_kernel(const float* __restrict__ E,
                                 __bf16* __restrict__ A,
                                 int B, int D) {
  const int w    = threadIdx.x >> 5;
  const int lane = threadIdx.x & 31;
  const int row  = blockIdx.x * 8 + w;
  if (row >= B) return;
  const float* p = E + (size_t)row * D;
  float ss = 0.f;
  for (int k = lane; k < D; k += 32) { float v = p[k]; ss += v * v; }
  ss += __shfl_xor(ss, 16);
  ss += __shfl_xor(ss, 8);
  ss += __shfl_xor(ss, 4);
  ss += __shfl_xor(ss, 2);
  ss += __shfl_xor(ss, 1);
  const float inv = 1.0f / fmaxf(sqrtf(ss), 1e-12f);
  __bf16* q = A + (size_t)row * D;
  for (int k = lane; k < D; k += 32) q[k] = (__bf16)(p[k] * inv);
}

// ---------------- kernel 3: bf16 WMMA GEMM with fused epilogue ----------------
// Block: 256 threads (8 waves). Tile: all M x 128 N, K = D (512).
// W tile converted fp32->bf16 once into LDS (NT loads: W is read exactly once
// globally). Output stores are non-temporal: 410 MB streamed once, keeping L2
// for the reused A matrix.
__global__ __launch_bounds__(256)
void arc_gemm_kernel(const float* __restrict__ W,
                     const __bf16* __restrict__ A,
                     const float* __restrict__ winv,
                     float* __restrict__ out,
                     int Brows, int C, int D) {
  __shared__ __attribute__((aligned(16))) __bf16 sB[TN * LDB];

  const int tid = threadIdx.x;
  const int NB  = blockIdx.x * TN;

  // ---- cooperative load: W[NB..NB+127][0..D) fp32 -> bf16 in LDS ----
  const int vecPerRow = D >> 2;             // float4 per row
  const int total     = TN * vecPerRow;
  for (int idx = tid; idx < total; idx += blockDim.x) {
    const int r    = idx / vecPerRow;
    const int c4   = idx - r * vecPerRow;
    int col  = NB + r;
    int scol = (col < C) ? col : (C - 1);   // clamp: garbage cols never stored
    const f32x4_t f = __builtin_nontemporal_load(
        (const f32x4_t*)(W + (size_t)scol * D + (size_t)c4 * 4));
    union { __bf16 h[4]; unsigned long long u64; } pk;
    pk.h[0] = (__bf16)f.x; pk.h[1] = (__bf16)f.y;
    pk.h[2] = (__bf16)f.z; pk.h[3] = (__bf16)f.w;
    *(unsigned long long*)&sB[r * LDB + c4 * 4] = pk.u64;
  }
  __syncthreads();

  const int wav  = tid >> 5;
  const int lane = tid & 31;
  const int mw   = wav & 1;                 // 2 M groups of 32 rows
  const int nw   = wav >> 1;                // 4 N groups of 32 cols
  const int h    = lane >> 4;               // lane half
  const int mn   = lane & 15;               // M (for A) / N (for B,D) index

  // B fragment base pointers (fixed across the whole M loop)
  const __bf16* bbase0 = &sB[(nw * 32 +      mn) * LDB + 8 * h];
  const __bf16* bbase1 = &sB[(nw * 32 + 16 + mn) * LDB + 8 * h];

  const int col0 = NB + nw * 32 + mn;
  const int col1 = col0 + 16;
  const float wv0 = (col0 < C) ? winv[col0] * ARC_S : 0.f;
  const float wv1 = (col1 < C) ? winv[col1] * ARC_S : 0.f;

  for (int mb = 0; mb < Brows; mb += 64) {
    const int MR0 = mb + mw * 32;
    const __bf16* abase0 = A + (size_t)(MR0 + mn) * D + 8 * h;
    const __bf16* abase1 = abase0 + (size_t)16 * D;

    v8f_t c00 = {}, c01 = {}, c10 = {}, c11 = {};

#pragma unroll 2
    for (int kb = 0; kb < D; kb += 32) {
      union { v16bf_t v; uint4 q[2]; } a0, a1, b0, b1;
      // A 16x32 bf16 frag: K = kb+8h+{0..7} and kb+16+8h+{0..7}
      a0.q[0] = *(const uint4*)(abase0 + kb);
      a0.q[1] = *(const uint4*)(abase0 + kb + 16);
      a1.q[0] = *(const uint4*)(abase1 + kb);
      a1.q[1] = *(const uint4*)(abase1 + kb + 16);
      // B 32x16 bf16 frag: K = kb+8h+{0..7} and kb+16+8h+{0..7}, column = mn
      b0.q[0] = *(const uint4*)(bbase0 + kb);
      b0.q[1] = *(const uint4*)(bbase0 + kb + 16);
      b1.q[0] = *(const uint4*)(bbase1 + kb);
      b1.q[1] = *(const uint4*)(bbase1 + kb + 16);

      c00 = __builtin_amdgcn_wmma_f32_16x16x32_bf16(false, a0.v, false, b0.v,
                                                    (short)0, c00, false, false);
      c01 = __builtin_amdgcn_wmma_f32_16x16x32_bf16(false, a0.v, false, b1.v,
                                                    (short)0, c01, false, false);
      c10 = __builtin_amdgcn_wmma_f32_16x16x32_bf16(false, a1.v, false, b0.v,
                                                    (short)0, c10, false, false);
      c11 = __builtin_amdgcn_wmma_f32_16x16x32_bf16(false, a1.v, false, b1.v,
                                                    (short)0, c11, false, false);
    }

    // ---- epilogue: D layout -> lane mn = column, VGPR v -> row (v + 8h) ----
    const int rbase0 = MR0 + 8 * h;         // rows for c0x
    const int rbase1 = MR0 + 16 + 8 * h;    // rows for c1x
#pragma unroll
    for (int v = 0; v < 8; ++v) {
      const size_t ro0 = (size_t)(rbase0 + v) * (size_t)C;
      const size_t ro1 = (size_t)(rbase1 + v) * (size_t)C;
      if (col0 < C) {
        __builtin_nontemporal_store(c00[v] * wv0, &out[ro0 + col0]);
        __builtin_nontemporal_store(c10[v] * wv0, &out[ro1 + col0]);
      }
      if (col1 < C) {
        __builtin_nontemporal_store(c01[v] * wv1, &out[ro0 + col1]);
        __builtin_nontemporal_store(c11[v] * wv1, &out[ro1 + col1]);
      }
    }
  }
}

// ---------------- kernel 4: ArcFace margin fix at label columns ----------------
__global__ void arc_fix_kernel(const int* __restrict__ labels,
                               float* __restrict__ out,
                               int Brows, int C) {
  const int r = blockIdx.x * blockDim.x + threadIdx.x;
  if (r >= Brows) return;
  const int lbl = labels[r];
  const size_t idx = (size_t)r * (size_t)C + (size_t)lbl;
  float c = out[idx] / ARC_S;
  c = fminf(fmaxf(c, -1.0f + ARC_EPS), 1.0f - ARC_EPS);
  out[idx] = cosf(acosf(c) + ARC_M) * ARC_S;
}

extern "C" void kernel_launch(void* const* d_in, const int* in_sizes, int n_in,
                              void* d_out, int out_size, void* d_ws, size_t ws_size,
                              hipStream_t stream) {
  const float* E      = (const float*)d_in[0];
  const float* W      = (const float*)d_in[1];
  const int*   labels = (const int*)d_in[2];
  float*       out    = (float*)d_out;

  const int Brows = in_sizes[2];            // 1024
  const int D     = in_sizes[0] / Brows;    // 512
  const int C     = in_sizes[1] / D;        // 100000

  // workspace: [A bf16 (B*D*2, 256-aligned)] [winv (C floats)]
  size_t aBytes = ((size_t)Brows * (size_t)D * 2 + 255) & ~(size_t)255;
  __bf16* Abf  = (__bf16*)d_ws;
  float*  winv = (float*)((char*)d_ws + aBytes);

  arc_wnorm_kernel<<<(C + 7) / 8, 256, 0, stream>>>(W, winv, C, D);
  arc_enorm_kernel<<<(Brows + 7) / 8, 256, 0, stream>>>(E, Abf, Brows, D);
  arc_gemm_kernel<<<(C + TN - 1) / TN, 256, 0, stream>>>(W, Abf, winv, out, Brows, C, D);
  arc_fix_kernel<<<(Brows + 255) / 256, 256, 0, stream>>>(labels, out, Brows, C);
}
